// NonLocalLayer_34127810134726
// MI455X (gfx1250) — matrix operand order repaired and checked by reading, not compiled, not run on hardware
//
#include <hip/hip_runtime.h>

// ---------------- problem constants ----------------
constexpr int Bb   = 8;
constexpr int CIN  = 256;
constexpr int COUT = 256;
constexpr int FF   = 512;
constexpr int NH   = 8;
constexpr int HD   = 64;              // FF / NH
constexpr int Nt   = 8 * 14 * 14;     // 1568 tokens (T*H*W)
constexpr float SCALE = 1.0f / 16.0f; // 1/sqrt(CIN)
constexpr float EPS   = 1e-5f;

typedef __attribute__((ext_vector_type(16))) __bf16 bf16x16;
typedef __attribute__((ext_vector_type(8)))  float  f32x8;

union FragU { bf16x16 v; uint4 q[2]; };
union Pack8 { __bf16 h[8]; uint4 q; };

#define DEVI __device__ __forceinline__

// A-operand fragment (16xK tile, 16-bit): lane holds row (lane&15);
// lanes>=16 shift K by 8; two 8-half runs at +0 and +16. (ISA 7.12.2)
DEVI bf16x16 load_a_frag(const __bf16* m, int ld, int row, int k0, int lane) {
  int kk = k0 + ((lane >> 4) << 3);
  const __bf16* p = m + (size_t)row * ld + kk;
  FragU f;
  f.q[0] = *reinterpret_cast<const uint4*>(p);
  f.q[1] = *reinterpret_cast<const uint4*>(p + 16);
  return f.v;
}
// B-operand fragment (Kx16 tile, 16-bit) read from the row-major transpose
// B^T[col][k]: lane holds col (lane&15); lanes>=16 shift K by 8;
// two 8-half runs at +0 and +16. (ISA 7.12.2)
DEVI bf16x16 load_b_frag(const __bf16* m, int ld, int row, int k0, int lane) {
  int kk = k0 + ((lane >> 4) << 3);
  const __bf16* p = m + (size_t)row * ld + kk;
  FragU f;
  f.q[0] = *reinterpret_cast<const uint4*>(p);
  f.q[1] = *reinterpret_cast<const uint4*>(p + 16);
  return f.v;
}
DEVI f32x8 wmma_bf16(bf16x16 a, bf16x16 b, f32x8 c) {
  return __builtin_amdgcn_wmma_f32_16x16x32_bf16(false, a, false, b,
                                                 (short)0, c, false, false);
}

// async 16B global -> LDS copy (per-lane addresses), tracked by ASYNCcnt
DEVI void async_copy_b128(unsigned lds_off, unsigned long long gaddr) {
  asm volatile("global_load_async_to_lds_b128 %0, %1, off"
               :: "v"(lds_off), "v"(gaddr)
               : "memory");
}
#define WAIT_ASYNC(n) asm volatile("s_wait_asynccnt " #n ::: "memory")
#define WAIT_DS0()    asm volatile("s_wait_dscnt 0" ::: "memory")

DEVI unsigned lds_offset_of(const void* p) {
  return (unsigned)(uintptr_t)p;  // shared aperture: low 32 bits = LDS offset
}

// ---------------- kernel 0a: weight f32 -> bf16 ----------------
__global__ void convert_f32_bf16(const float* __restrict__ src,
                                 __bf16* __restrict__ dst, int n) {
  int i = blockIdx.x * blockDim.x + threadIdx.x;
  if (i < n) dst[i] = (__bf16)src[i];
}

// ---------------- kernel 0b: x[b][c][n] f32 -> xb[b][n][c] bf16 ----------------
__global__ __launch_bounds__(256) void transpose_x_kernel(
    const float* __restrict__ x, __bf16* __restrict__ xb) {
  __shared__ float tile[32][33];
  int b  = blockIdx.z;
  int c0 = blockIdx.y * 32;
  int n0 = blockIdx.x * 32;
  int tx = threadIdx.x, ty = threadIdx.y;
#pragma unroll
  for (int i = 0; i < 4; ++i) {
    int cy = ty + i * 8;
    tile[cy][tx] = x[((size_t)b * CIN + (c0 + cy)) * Nt + (n0 + tx)];
  }
  __syncthreads();
#pragma unroll
  for (int i = 0; i < 4; ++i) {
    int ny = ty + i * 8;
    xb[((size_t)b * Nt + (n0 + ny)) * CIN + (c0 + tx)] = (__bf16)tile[tx][ny];
  }
}

// ---------------- kernel 1: QKV projection ----------------
__global__ __launch_bounds__(32) void proj_qkv_kernel(
    const __bf16* __restrict__ xb, const __bf16* __restrict__ Wk,
    const __bf16* __restrict__ Wq, const __bf16* __restrict__ Wv,
    __bf16* __restrict__ Kt, __bf16* __restrict__ Qt, __bf16* __restrict__ Vc) {
  int lane = threadIdx.x;
  int r    = lane & 15;
  int hi8  = (lane >> 4) << 3;
  int b    = blockIdx.z;
  int fb   = blockIdx.y * 16;
  int nb   = blockIdx.x * 16;
  const __bf16* xrow = xb + (size_t)b * Nt * CIN;

  f32x8 cK = {}, cQ = {}, cV = {};
  for (int k = 0; k < CIN; k += 32) {
    if (k + 32 < CIN) __builtin_prefetch(xrow + (size_t)(nb + r) * CIN + k + 32, 0, 1);
    bf16x16 bx = load_b_frag(xrow, CIN, nb + r, k, lane); // x^T as B-operand
    bf16x16 aK = load_a_frag(Wk, CIN, fb + r, k, lane);
    bf16x16 aQ = load_a_frag(Wq, CIN, fb + r, k, lane);
    cK = wmma_bf16(aK, bx, cK);
    cQ = wmma_bf16(aQ, bx, cQ);
    bf16x16 aX = load_a_frag(xrow, CIN, nb + r, k, lane); // x as A-operand
    bf16x16 bW = load_b_frag(Wv, CIN, fb + r, k, lane);   // W^T as B-operand
    cV = wmma_bf16(aX, bW, cV);
  }
  int h  = fb >> 6;
  int cb = fb & 63;
  {  // K/Q: lane has col n = nb+r, rows f = fb+hi8+rr -> 8 consecutive c, b128 store
    Pack8 pk, pq;
#pragma unroll
    for (int rr = 0; rr < 8; ++rr) { pk.h[rr] = (__bf16)cK[rr]; pq.h[rr] = (__bf16)cQ[rr]; }
    size_t idx = (((size_t)b * NH + h) * Nt + (nb + r)) * HD + cb + hi8;
    *reinterpret_cast<uint4*>(Kt + idx) = pk.q;
    *reinterpret_cast<uint4*>(Qt + idx) = pq.q;
  }
  {  // V: lane has col f = fb+r -> c = cb+r, rows n = nb+hi8+rr -> 8 consecutive n
    Pack8 pv;
#pragma unroll
    for (int rr = 0; rr < 8; ++rr) pv.h[rr] = (__bf16)cV[rr];
    size_t idx = (((size_t)b * NH + h) * HD + (cb + r)) * Nt + nb + hi8;
    *reinterpret_cast<uint4*>(Vc + idx) = pv.q;
  }
}

// ---------------- kernel 2: flash attention ----------------
// 2 waves per workgroup: wave w owns i-tile (blockIdx.x*32 + w*16).
// Q/V j-tiles are staged cooperatively into LDS with async b128 copies,
// double-buffered; each wave issues exactly 8 async instructions per tile.
__global__ __launch_bounds__(64) void flash_attn_kernel(
    const __bf16* __restrict__ Kt, const __bf16* __restrict__ Qt,
    const __bf16* __restrict__ Vc, __bf16* __restrict__ Agg) {
  __shared__ __align__(16) __bf16 Qs[2][32 * 64];  // [buf][j_local][c]
  __shared__ __align__(16) __bf16 Vs[2][64 * 32];  // [buf][c][j_local]
  __shared__ __align__(16) __bf16 Pbuf[2][16 * 32];
  __shared__ float rowBc[2][16];
  int tid  = threadIdx.x;
  int wave = tid >> 5;
  int lane = tid & 31;
  int r    = lane & 15;
  int hi8  = (lane >> 4) << 3;
  int ib   = blockIdx.x * 32 + wave * 16;
  int h    = blockIdx.y;
  int b    = blockIdx.z;
  size_t bh = (size_t)b * NH + h;
  const __bf16* Kb = Kt + bh * Nt * HD;
  const __bf16* Qb = Qt + bh * Nt * HD;
  const __bf16* Vb = Vc + bh * HD * Nt;
  __bf16* Pw = &Pbuf[wave][0];
  float*  Rw = &rowBc[wave][0];

  // K rows for this wave's i-tile stay resident in registers
  bf16x16 aK0 = load_a_frag(Kb, HD, ib + r, 0, lane);
  bf16x16 aK1 = load_a_frag(Kb, HD, ib + r, 32, lane);

  // issue the 8 async b128 copies this wave owns for tile starting at jb
  auto issue_tile = [&](int buf, int jb) {
    // Q: rows jb..jb+31 are one contiguous 4KB block; this wave moves 2KB.
    {
      unsigned long long g = (unsigned long long)(uintptr_t)Qb +
                             (unsigned long long)jb * (HD * 2) + wave * 2048;
      unsigned l = lds_offset_of(&Qs[buf][0]) + wave * 2048;
#pragma unroll
      for (int i = 0; i < 4; ++i) {
        unsigned o = i * 512 + lane * 16;
        async_copy_b128(l + o, g + o);
      }
    }
    // V: 64 rows x 64B (row c, cols jb..jb+31); this wave moves 32 rows.
    {
#pragma unroll
      for (int i = 0; i < 4; ++i) {
        int row = wave * 32 + i * 8 + (lane >> 2);
        unsigned intra = (lane & 3) * 16;
        unsigned long long g = (unsigned long long)(uintptr_t)Vb +
                               ((unsigned long long)row * Nt + jb) * 2 + intra;
        unsigned l = lds_offset_of(&Vs[buf][0]) + row * 64 + intra;
        async_copy_b128(l, g);
      }
    }
  };

  f32x8 acc0 = {}, acc1 = {}, acc2 = {}, acc3 = {};
  float m[8], s[8];
#pragma unroll
  for (int rr = 0; rr < 8; ++rr) { m[rr] = -3.0e38f; s[rr] = 0.0f; }

  issue_tile(0, 0);
  constexpr int NTILES = Nt / 32;  // 49
  for (int t = 0; t < NTILES; ++t) {
    int cur = t & 1;
    if (t + 1 < NTILES) {
      issue_tile(1 - cur, (t + 1) * 32);
      WAIT_ASYNC(8);   // 8 newer outstanding -> tile t's copies complete (in order)
    } else {
      WAIT_ASYNC(0);
    }
    __syncthreads();   // tile t visible to both waves
    const __bf16* Qsc = &Qs[cur][0];
    const __bf16* Vsc = &Vs[cur][0];

    // scores S[i, j] = sum_c K[i,c] * Q[j,c]  (Q fragments from LDS)
    f32x8 s0 = {}, s1 = {};
    {
      bf16x16 bq;
      bq = load_b_frag(Qsc, HD, r, 0, lane);       s0 = wmma_bf16(aK0, bq, s0);
      bq = load_b_frag(Qsc, HD, r, 32, lane);      s0 = wmma_bf16(aK1, bq, s0);
      bq = load_b_frag(Qsc, HD, 16 + r, 0, lane);  s1 = wmma_bf16(aK0, bq, s1);
      bq = load_b_frag(Qsc, HD, 16 + r, 32, lane); s1 = wmma_bf16(aK1, bq, s1);
    }
    // row-max across the 16-lane group that owns each row
    float rm[8];
#pragma unroll
    for (int rr = 0; rr < 8; ++rr) rm[rr] = fmaxf(s0[rr], s1[rr]);
    for (int msk = 1; msk < 16; msk <<= 1)
#pragma unroll
      for (int rr = 0; rr < 8; ++rr) rm[rr] = fmaxf(rm[rr], __shfl_xor(rm[rr], msk, 32));

    float alpha[8], psum[8];
    f32x8 p0, p1;
#pragma unroll
    for (int rr = 0; rr < 8; ++rr) {
      float mnew = fmaxf(m[rr], rm[rr] * SCALE);
      alpha[rr]  = __expf(m[rr] - mnew);
      m[rr]      = mnew;
      float e0 = __expf(s0[rr] * SCALE - mnew);
      float e1 = __expf(s1[rr] * SCALE - mnew);
      p0[rr] = e0; p1[rr] = e1;
      psum[rr] = e0 + e1;
    }
    for (int msk = 1; msk < 16; msk <<= 1)
#pragma unroll
      for (int rr = 0; rr < 8; ++rr) psum[rr] += __shfl_xor(psum[rr], msk, 32);
#pragma unroll
    for (int rr = 0; rr < 8; ++rr) s[rr] = s[rr] * alpha[rr] + psum[rr];

    // P -> per-wave LDS tile [i][32 j]; alpha broadcast via LDS
#pragma unroll
    for (int rr = 0; rr < 8; ++rr) {
      Pw[(hi8 + rr) * 32 + r]      = (__bf16)p0[rr];
      Pw[(hi8 + rr) * 32 + r + 16] = (__bf16)p1[rr];
    }
    if (r == 0) {
#pragma unroll
      for (int rr = 0; rr < 8; ++rr) Rw[hi8 + rr] = alpha[rr];
    }
    WAIT_DS0();
    float av = Rw[r];  // alpha for accumulator column i = r
#pragma unroll
    for (int rr = 0; rr < 8; ++rr) {
      acc0[rr] *= av; acc1[rr] *= av; acc2[rr] *= av; acc3[rr] *= av;
    }
    // apply: aggT[c, i] += V[c, j] * P[i, j]  (V and P fragments from LDS)
    bf16x16 bp = load_b_frag(Pw, 32, r, 0, lane);
    bf16x16 a0 = load_a_frag(Vsc, 32, 0  + r, 0, lane);
    bf16x16 a1 = load_a_frag(Vsc, 32, 16 + r, 0, lane);
    bf16x16 a2 = load_a_frag(Vsc, 32, 32 + r, 0, lane);
    bf16x16 a3 = load_a_frag(Vsc, 32, 48 + r, 0, lane);
    acc0 = wmma_bf16(a0, bp, acc0);
    acc1 = wmma_bf16(a1, bp, acc1);
    acc2 = wmma_bf16(a2, bp, acc2);
    acc3 = wmma_bf16(a3, bp, acc3);
    __syncthreads();   // both waves done reading buf before it is refilled
  }

  // final 1/sum scaling (per column i = r), broadcast via per-wave LDS
  if (r == 0) {
#pragma unroll
    for (int rr = 0; rr < 8; ++rr) Rw[hi8 + rr] = s[rr];
  }
  WAIT_DS0();
  float inv = 1.0f / Rw[r];

  // write Agg[b][n][f]: lane owns token n = ib + r; rows are 8 consecutive f
  size_t nrow = ((size_t)b * Nt + ib + r) * FF + (size_t)h * HD;
  Pack8 pa;
#pragma unroll
  for (int rr = 0; rr < 8; ++rr) pa.h[rr] = (__bf16)(acc0[rr] * inv);
  *reinterpret_cast<uint4*>(Agg + nrow + 0  + hi8) = pa.q;
#pragma unroll
  for (int rr = 0; rr < 8; ++rr) pa.h[rr] = (__bf16)(acc1[rr] * inv);
  *reinterpret_cast<uint4*>(Agg + nrow + 16 + hi8) = pa.q;
#pragma unroll
  for (int rr = 0; rr < 8; ++rr) pa.h[rr] = (__bf16)(acc2[rr] * inv);
  *reinterpret_cast<uint4*>(Agg + nrow + 32 + hi8) = pa.q;
#pragma unroll
  for (int rr = 0; rr < 8; ++rr) pa.h[rr] = (__bf16)(acc3[rr] * inv);
  *reinterpret_cast<uint4*>(Agg + nrow + 48 + hi8) = pa.q;
}

// ---------------- kernel 3: output projection + bias + relu + skip ----------------
__global__ __launch_bounds__(32) void out_proj_kernel(
    const __bf16* __restrict__ Wob, const __bf16* __restrict__ Agg,
    const float* __restrict__ x, const float* __restrict__ bout,
    float* __restrict__ y) {
  int lane = threadIdx.x;
  int r    = lane & 15;
  int hi8  = (lane >> 4) << 3;
  int b    = blockIdx.z;
  int ob   = blockIdx.y * 16;
  int nb   = blockIdx.x * 16;
  const __bf16* arow = Agg + (size_t)b * Nt * FF;

  f32x8 c = {};
  for (int k = 0; k < FF; k += 32) {
    bf16x16 a  = load_a_frag(Wob, FF, ob + r, k, lane);
    bf16x16 bb = load_b_frag(arow, FF, nb + r, k, lane);
    c = wmma_bf16(a, bb, c);
  }
  int n = nb + r;
#pragma unroll
  for (int rr = 0; rr < 8; ++rr) {
    int o = ob + hi8 + rr;
    float v = fmaxf(c[rr] + bout[o], 0.0f);
    size_t idx = ((size_t)b * COUT + o) * Nt + n;
    y[idx] = x[idx] + v;
  }
}

// ---------------- kernel 4: BN stats per channel over (B, N) ----------------
__global__ __launch_bounds__(256) void bn_stats_kernel(
    const float* __restrict__ y, float* __restrict__ mean, float* __restrict__ istd) {
  int o = blockIdx.x;
  int t = threadIdx.x;
  float s = 0.0f, s2 = 0.0f;
  for (int b = 0; b < Bb; ++b) {
    const float* p = y + ((size_t)b * COUT + o) * Nt;
    for (int n = t; n < Nt; n += 256) { float v = p[n]; s += v; s2 += v * v; }
  }
  __shared__ float sh[256], sh2[256];
  sh[t] = s; sh2[t] = s2;
  __syncthreads();
  for (int st = 128; st > 0; st >>= 1) {
    if (t < st) { sh[t] += sh[t + st]; sh2[t] += sh2[t + st]; }
    __syncthreads();
  }
  if (t == 0) {
    const float invc = 1.0f / (float)(Bb * Nt);
    float mu  = sh[0] * invc;
    float var = sh2[0] * invc - mu * mu;   // biased variance, like torch BN
    mean[o] = mu;
    istd[o] = rsqrtf(var + EPS);
  }
}

// ---------------- kernel 5: BN apply ----------------
__global__ __launch_bounds__(256) void bn_apply_kernel(
    const float* __restrict__ y, const float* __restrict__ mean,
    const float* __restrict__ istd, const float* __restrict__ gamma,
    const float* __restrict__ beta, float* __restrict__ out, int total) {
  int i = blockIdx.x * blockDim.x + threadIdx.x;
  if (i >= total) return;
  int o = (i / Nt) % COUT;
  out[i] = (y[i] - mean[o]) * istd[o] * gamma[o] + beta[o];
}

// ---------------- host launcher ----------------
extern "C" void kernel_launch(void* const* d_in, const int* in_sizes, int n_in,
                              void* d_out, int out_size, void* d_ws, size_t ws_size,
                              hipStream_t stream) {
  (void)in_sizes; (void)n_in; (void)out_size; (void)ws_size;
  const float* x     = (const float*)d_in[0];
  const float* WK    = (const float*)d_in[1];
  const float* WQ    = (const float*)d_in[2];
  const float* WV    = (const float*)d_in[3];
  const float* Wout  = (const float*)d_in[4];
  const float* bout  = (const float*)d_in[5];
  const float* gamma = (const float*)d_in[6];
  const float* beta  = (const float*)d_in[7];

  char* base = (char*)d_ws;
  size_t off = 0;
  auto take = [&](size_t bytes) -> char* {
    off = (off + 255) & ~(size_t)255;
    char* p = base + off;
    off += bytes;
    return p;
  };
  __bf16* wkb = (__bf16*)take((size_t)FF * CIN * 2);
  __bf16* wqb = (__bf16*)take((size_t)FF * CIN * 2);
  __bf16* wvb = (__bf16*)take((size_t)FF * CIN * 2);
  __bf16* wob = (__bf16*)take((size_t)COUT * FF * 2);
  __bf16* xb  = (__bf16*)take((size_t)Bb * Nt * CIN * 2);
  __bf16* Ktb = (__bf16*)take((size_t)Bb * NH * Nt * HD * 2);
  __bf16* Qtb = (__bf16*)take((size_t)Bb * NH * Nt * HD * 2);
  __bf16* Vcb = (__bf16*)take((size_t)Bb * NH * HD * Nt * 2);
  __bf16* Agg = (__bf16*)take((size_t)Bb * Nt * FF * 2);
  float*  y   = (float*)take((size_t)Bb * COUT * Nt * 4);
  float*  mean= (float*)take((size_t)COUT * 4);
  float*  istd= (float*)take((size_t)COUT * 4);

  convert_f32_bf16<<<(FF * CIN + 255) / 256, 256, 0, stream>>>(WK, wkb, FF * CIN);
  convert_f32_bf16<<<(FF * CIN + 255) / 256, 256, 0, stream>>>(WQ, wqb, FF * CIN);
  convert_f32_bf16<<<(FF * CIN + 255) / 256, 256, 0, stream>>>(WV, wvb, FF * CIN);
  convert_f32_bf16<<<(COUT * FF + 255) / 256, 256, 0, stream>>>(Wout, wob, COUT * FF);

  transpose_x_kernel<<<dim3(Nt / 32, CIN / 32, Bb), dim3(32, 8), 0, stream>>>(x, xb);
  proj_qkv_kernel<<<dim3(Nt / 16, FF / 16, Bb), 32, 0, stream>>>(xb, wkb, wqb, wvb,
                                                                 Ktb, Qtb, Vcb);
  flash_attn_kernel<<<dim3(Nt / 32, NH, Bb), 64, 0, stream>>>(Ktb, Qtb, Vcb, Agg);
  out_proj_kernel<<<dim3(Nt / 16, COUT / 16, Bb), 32, 0, stream>>>(wob, Agg, x, bout, y);
  bn_stats_kernel<<<COUT, 256, 0, stream>>>(y, mean, istd);

  int total = Bb * COUT * Nt;
  bn_apply_kernel<<<(total + 255) / 256, 256, 0, stream>>>(y, mean, istd, gamma, beta,
                                                           (float*)d_out, total);
}
